// TtShiftedWindowAttention_64287070487324
// MI455X (gfx1250) — matrix operand order repaired and checked by reading, not compiled, not run on hardware
//
#include <hip/hip_runtime.h>
#include <hip/hip_bf16.h>

typedef __attribute__((ext_vector_type(16))) __bf16 v16bf;
typedef __attribute__((ext_vector_type(8)))  float  v8f;
typedef __attribute__((ext_vector_type(4)))  unsigned int u32x4;
typedef __attribute__((ext_vector_type(8)))  int  i32x8;
typedef __attribute__((ext_vector_type(4)))  int  i32x4;

#define WIN     7
#define N_POS   49
#define HEADS   6
#define HD      32
#define C_DIM   192
#define IMG     56
#define SHIFT   3
#define BW      2048          // B(32) * nW(64)
#define LDA_PAD 200

#if __has_builtin(__builtin_amdgcn_tensor_load_to_lds)
#  define HAVE_TDM 1
#  if __has_include(<hip/amd_detail/amd_gfx1250_TDM.h>)
#    define TDM_6ARG 1       // therock-10.0 headers -> 6-arg builtin
#  else
#    define TDM_6ARG 0       // ROCm 7.2 -> 5-arg builtin
#  endif
#else
#  define HAVE_TDM 0
#endif

// ---- fragment loaders (ISA 7.12.2 layouts) -------------------------------

// A-matrix 16x32 bf16 (row-major source, stride lda elements).
// lane 0-15: M=lane, K = {0..7, 16..23}; lane 16-31: same M rows, K +8.
static __device__ __forceinline__ v16bf frag_a_rm(const __bf16* s, int lda,
                                                  int mbase, int kbase, int lane) {
  const int m  = mbase + (lane & 15);
  const int hi = (lane >> 4) & 1;
  const __bf16* p = s + (size_t)m * lda + kbase + hi * 8;
  v16bf f;
#pragma unroll
  for (int e = 0; e < 8; ++e) f[e] = p[e];
#pragma unroll
  for (int e = 0; e < 8; ++e) f[8 + e] = p[16 + e];
  return f;
}

// B-matrix 32x16 where B[k][n] = S[n][k] (S row-major bf16, stride ld).
// lane 0-15: n=lane, K = {0..7, 16..23}; lane 16-31: n=lane-16, K +8.
static __device__ __forceinline__ v16bf frag_b_rowT_bf16(const __bf16* s, int ld,
                                                         int kbase, int nbase, int lane) {
  const int n  = nbase + (lane & 15);
  const int hi = (lane >> 4) & 1;
  const __bf16* p = s + (size_t)n * ld + kbase + hi * 8;
  v16bf f;
#pragma unroll
  for (int e = 0; e < 8; ++e) f[e] = p[e];
#pragma unroll
  for (int e = 0; e < 8; ++e) f[8 + e] = p[16 + e];
  return f;
}

static __device__ __forceinline__ v8f wmma_bf16(v16bf a, v16bf b, v8f c) {
  return __builtin_amdgcn_wmma_f32_16x16x32_bf16(false, a, false, b, (short)0, c,
                                                 false, false);
}

// ---- kernel 0: one-shot weight f32 -> bf16 conversion --------------------
__global__ void __launch_bounds__(256)
cvt_weights(const float* __restrict__ qw, const float* __restrict__ pw,
            __bf16* __restrict__ qwb, __bf16* __restrict__ pwb) {
  const int i = blockIdx.x * 256 + threadIdx.x;
  if (i < 3 * C_DIM * C_DIM) qwb[i] = (__bf16)qw[i];
  if (i < C_DIM * C_DIM)     pwb[i] = (__bf16)pw[i];
}

// ---- kernel 1: fused shifted-window gather + QKV GEMM --------------------
__global__ void __launch_bounds__(256)
qkv_kernel(const float* __restrict__ x, const __bf16* __restrict__ w,
           const float* __restrict__ bias,
           __bf16* __restrict__ qB, __bf16* __restrict__ kB, __bf16* __restrict__ vB) {
  __shared__ __bf16 As[64 * LDA_PAD];
  const int wb  = blockIdx.x;
  const int b   = wb >> 6;
  const int win = wb & 63;
  const int wy  = win >> 3, wx = win & 7;

  // stage shifted-gathered window tile (49x192, padded to 64 rows) as bf16
  for (int idx = threadIdx.x; idx < 64 * C_DIM; idx += 256) {
    const int row = idx / C_DIM, col = idx % C_DIM;
    float vl = 0.f;
    if (row < N_POS) {
      const int ny = row / WIN, nx = row % WIN;
      const int hh = (wy * WIN + ny + SHIFT) % IMG;   // roll(-SHIFT) gather
      const int ww = (wx * WIN + nx + SHIFT) % IMG;
      vl = x[(((size_t)b * IMG + hh) * IMG + ww) * C_DIM + col];
    }
    As[row * LDA_PAD + col] = (__bf16)vl;
  }
  __syncthreads();

  const int lane = threadIdx.x & 31;
  const int wv   = threadIdx.x >> 5;
  const int hi   = lane >> 4;
  const int cl   = lane & 15;

  for (int t = wv; t < 4 * 36; t += 8) {          // 64x576 output, 16x16 tiles
    const int mt = t / 36, nt = t % 36;
    __builtin_prefetch(w + (size_t)(nt * 16 + cl) * C_DIM, 0, 0);
    v8f acc = {0.f, 0.f, 0.f, 0.f, 0.f, 0.f, 0.f, 0.f};
#pragma unroll
    for (int kk = 0; kk < 6; ++kk) {
      v16bf a  = frag_a_rm(As, LDA_PAD, mt * 16, kk * 32, lane);
      v16bf bf = frag_b_rowT_bf16(w, C_DIM, kk * 32, nt * 16, lane);
      acc = wmma_bf16(a, bf, acc);
    }
    // tile-invariant epilogue terms hoisted out of the row loop
    const int   o   = nt * 16 + cl;
    const float bo  = bias[o];                     // one unconditional load
    const int   sec = o / C_DIM;                   // 0=q 1=k 2=v (tile-uniform)
    const int   h   = (o >> 5) % HEADS;
    const int   d   = o & 31;
    const size_t task = (size_t)wb * HEADS + h;
    __bf16* qp = qB + (task * 64) * 32 + d;
    __bf16* kp = kB + (task * 64) * 32 + d;
    __bf16* vp = vB + (task * 32 + d) * 64;
#pragma unroll
    for (int r = 0; r < 8; ++r) {
      const int row = mt * 16 + (hi ? r + 8 : r);
      const float val = (row < N_POS) ? acc[r] + bo : 0.f;
      if (sec == 0)
        qp[(size_t)row * 32] = (__bf16)(val * 0.17677669529663687f);
      else if (sec == 1)
        kp[(size_t)row * 32] = (__bf16)val;
      else
        vp[row] = (__bf16)val;                     // V stored transposed
    }
  }
}

// ---- kernel 2: per-(window,head) attention, one wave per task ------------
__global__ void __launch_bounds__(128)
attn_kernel(const __bf16* __restrict__ qB, const __bf16* __restrict__ kB,
            const __bf16* __restrict__ vB, const float* __restrict__ rel_table,
            __bf16* __restrict__ aB) {
  __shared__ __bf16 P[4][16 * 64];                 // per-wave P row-block
  const int lane = threadIdx.x & 31;
  const int wv   = threadIdx.x >> 5;
  const int task = blockIdx.x * 4 + wv;            // 12288 tasks exactly
  const int wb   = task / HEADS;
  const int h    = task % HEADS;
  const int win  = wb & 63;
  const int wy   = win >> 3, wx = win & 7;

  const __bf16* q = qB + (size_t)task * 64 * 32;
  const __bf16* k = kB + (size_t)task * 64 * 32;
  const __bf16* v = vB + (size_t)task * 32 * 64;
  __bf16* aOut = aB + (size_t)wb * 64 * C_DIM + h * HD;

  // preload K^T and V B-fragments (reused across all 4 query row-blocks)
  v16bf bk[4], bv[2][2];
#pragma unroll
  for (int nt = 0; nt < 4; ++nt) bk[nt] = frag_b_rowT_bf16(k, 32, 0, nt * 16, lane);
#pragma unroll
  for (int ks = 0; ks < 2; ++ks)
#pragma unroll
    for (int on = 0; on < 2; ++on)
      bv[ks][on] = frag_b_rowT_bf16(v, 64, ks * 32, on * 16, lane);

  const int hi = lane >> 4;
  const int cl = lane & 15;

  for (int mt = 0; mt < 4; ++mt) {
    // S row-block = q[mt] @ K^T  (K-dim = head_dim = 32 -> one WMMA each)
    v16bf aq = frag_a_rm(q, 32, mt * 16, 0, lane);
    v8f s[4];
#pragma unroll
    for (int nt = 0; nt < 4; ++nt) {
      v8f z = {0.f, 0.f, 0.f, 0.f, 0.f, 0.f, 0.f, 0.f};
      s[nt] = wmma_bf16(aq, bk[nt], z);
    }
    // analytic relative-position bias + shift mask, C-layout registers
#pragma unroll
    for (int r = 0; r < 8; ++r) {
      const int gm = mt * 16 + (hi ? r + 8 : r);
      const int ny = gm / WIN, nx = gm % WIN;
      const int hr = wy * WIN + ny, wr = wx * WIN + nx;
      const int regN = ((hr < 49) ? 0 : (hr < 53) ? 1 : 2) * 3 +
                       ((wr < 49) ? 0 : (wr < 53) ? 1 : 2);
#pragma unroll
      for (int nt = 0; nt < 4; ++nt) {
        const int gn = nt * 16 + cl;
        float val = s[nt][r];
        if (gm < N_POS && gn < N_POS) {
          const int my = gn / WIN, mx = gn % WIN;
          val += rel_table[((ny - my + 6) * 13 + (nx - mx + 6)) * HEADS + h];
          const int hc = wy * WIN + my, wc = wx * WIN + mx;
          const int regM = ((hc < 49) ? 0 : (hc < 53) ? 1 : 2) * 3 +
                           ((wc < 49) ? 0 : (wc < 53) ? 1 : 2);
          if (regN != regM) val -= 100.f;
        } else {
          val = -1e30f;                            // mask padded key columns
        }
        s[nt][r] = val;
      }
    }
    // row softmax: 4 elems/lane + 16-lane half-wave shuffle reduction
#pragma unroll
    for (int r = 0; r < 8; ++r) {
      float mx = fmaxf(fmaxf(s[0][r], s[1][r]), fmaxf(s[2][r], s[3][r]));
      for (int d = 1; d < 16; d <<= 1) mx = fmaxf(mx, __shfl_xor(mx, d, 32));
      float e0 = __expf(s[0][r] - mx), e1 = __expf(s[1][r] - mx);
      float e2 = __expf(s[2][r] - mx), e3 = __expf(s[3][r] - mx);
      float sum = e0 + e1 + e2 + e3;
      for (int d = 1; d < 16; d <<= 1) sum += __shfl_xor(sum, d, 32);
      const float inv = 1.f / sum;
      const int lrow = hi ? r + 8 : r;
      P[wv][lrow * 64 +  0 + cl] = (__bf16)(e0 * inv);
      P[wv][lrow * 64 + 16 + cl] = (__bf16)(e1 * inv);
      P[wv][lrow * 64 + 32 + cl] = (__bf16)(e2 * inv);
      P[wv][lrow * 64 + 48 + cl] = (__bf16)(e3 * inv);
    }
    // O row-block = P @ V  (K-dim 64 -> 2 WMMA steps)
    v16bf ap[2];
#pragma unroll
    for (int ks = 0; ks < 2; ++ks) ap[ks] = frag_a_rm(&P[wv][0], 64, 0, ks * 32, lane);
#pragma unroll
    for (int on = 0; on < 2; ++on) {
      v8f o = {0.f, 0.f, 0.f, 0.f, 0.f, 0.f, 0.f, 0.f};
#pragma unroll
      for (int ks = 0; ks < 2; ++ks) o = wmma_bf16(ap[ks], bv[ks][on], o);
#pragma unroll
      for (int r = 0; r < 8; ++r) {
        const int gm = mt * 16 + (hi ? r + 8 : r);
        const float val = (gm < N_POS) ? o[r] : 0.f;
        aOut[(size_t)gm * C_DIM + on * 16 + cl] = (__bf16)val;
      }
    }
  }
}

// ---- kernel 3: output projection + reverse window/roll scatter -----------
// A tile stage uses the Tensor Data Mover: one contiguous 24576-byte
// 1-D transfer (bf16 [64][192] window of attn_buf) into LDS offset 0.
__global__ void __launch_bounds__(256)
proj_kernel(const __bf16* __restrict__ aB, const __bf16* __restrict__ w,
            const float* __restrict__ bias, float* __restrict__ out) {
  __shared__ __bf16 As[64 * C_DIM];                // only shared var -> LDS off 0
  const int wb  = blockIdx.x;
  const int b   = wb >> 6;
  const int win = wb & 63;
  const int wy  = win >> 3, wx = win & 7;

#if HAVE_TDM
  if (threadIdx.x < 32) {                          // single wave issues the DMA
    const unsigned long long ga =
        (unsigned long long)(uintptr_t)(aB + (size_t)wb * 64 * C_DIM);
    // D# group0: count=1, lds_addr=0, global_addr, type=2 (ISA 8.3)
    u32x4 g0;
    g0[0] = 1u;                                    // count=1, user mode
    g0[1] = 0u;                                    // lds_addr = 0
    g0[2] = (unsigned)ga;                          // global_addr[31:0]
    g0[3] = (unsigned)((ga >> 32) & 0x01FFFFFFu) | (2u << 30);  // [56:32]|type
    // D# group1: data_size=4B, tensor_dim0=tile_dim0=6144, 1-D (ISA 8.4)
    i32x8 g1;
    g1[0] = 2 << 16;                               // data_size = 4 bytes
    g1[1] = 6144 << 16;                            // tensor_dim0[15:0] @bit48
    g1[2] = 1 << 16;                               // tensor_dim0 hi=0, tensor_dim1=1
    g1[3] = 6144 << 16;                            // tile_dim0 @bit112
    g1[4] = 0;                                     // tile_dim1=0 (1-D), tile_dim2=0
    g1[5] = 6144;                                  // tensor_dim0_stride lo
    g1[6] = 0;
    g1[7] = 0;
    i32x4 z4 = {0, 0, 0, 0};
#if TDM_6ARG
    i32x8 z8 = {0, 0, 0, 0, 0, 0, 0, 0};
    __builtin_amdgcn_tensor_load_to_lds(g0, g1, z4, z4, z8, 0);
#else
    __builtin_amdgcn_tensor_load_to_lds(g0, g1, z4, z4, 0);
#endif
    __builtin_amdgcn_s_wait_tensorcnt(0);
  }
  __syncthreads();
#else
  for (int idx = threadIdx.x; idx < 64 * C_DIM; idx += 256)
    As[idx] = aB[(size_t)wb * 64 * C_DIM + idx];
  __syncthreads();
#endif

  const int lane = threadIdx.x & 31;
  const int wv   = threadIdx.x >> 5;
  const int hi   = lane >> 4;
  const int cl   = lane & 15;

  for (int t = wv; t < 4 * 12; t += 8) {          // 64x192 output tiles
    const int mt = t / 12, nt = t % 12;
    __builtin_prefetch(w + (size_t)(nt * 16 + cl) * C_DIM, 0, 0);
    v8f acc = {0.f, 0.f, 0.f, 0.f, 0.f, 0.f, 0.f, 0.f};
#pragma unroll
    for (int kk = 0; kk < 6; ++kk) {
      v16bf a  = frag_a_rm(As, C_DIM, mt * 16, kk * 32, lane);
      v16bf bf = frag_b_rowT_bf16(w, C_DIM, kk * 32, nt * 16, lane);
      acc = wmma_bf16(a, bf, acc);
    }
    const int   o  = nt * 16 + cl;
    const float bo = bias[o];                      // hoisted, unconditional
    float* outc = out + (size_t)b * IMG * IMG * C_DIM + o;
#pragma unroll
    for (int r = 0; r < 8; ++r) {
      const int row = mt * 16 + (hi ? r + 8 : r);
      if (row < N_POS) {
        const int ny = row / WIN, nx = row % WIN;
        const int hh = (wy * WIN + ny + SHIFT) % IMG;  // roll(+SHIFT) scatter
        const int ww = (wx * WIN + nx + SHIFT) % IMG;
        outc[((size_t)hh * IMG + ww) * C_DIM] = acc[r] + bo;
      }
    }
  }
}

// ---- launcher ------------------------------------------------------------
extern "C" void kernel_launch(void* const* d_in, const int* in_sizes, int n_in,
                              void* d_out, int out_size, void* d_ws, size_t ws_size,
                              hipStream_t stream) {
  (void)in_sizes; (void)n_in; (void)out_size; (void)ws_size;
  const float* x      = (const float*)d_in[0];
  const float* qkv_w  = (const float*)d_in[1];
  const float* qkv_b  = (const float*)d_in[2];
  const float* proj_w = (const float*)d_in[3];
  const float* proj_b = (const float*)d_in[4];
  const float* rel    = (const float*)d_in[5];

  const size_t per = (size_t)BW * HEADS * 64 * 32;   // 25,165,824 bf16 elems
  __bf16* qB  = (__bf16*)d_ws;
  __bf16* kB  = qB + per;
  __bf16* vB  = kB + per;
  __bf16* aB  = vB + per;                            // [BW][64][192] bf16
  __bf16* wqB = aB + (size_t)BW * 64 * C_DIM;        // 110592 bf16
  __bf16* wpB = wqB + 3 * C_DIM * C_DIM;             // 36864 bf16

  cvt_weights<<<(3 * C_DIM * C_DIM + 255) / 256, 256, 0, stream>>>(qkv_w, proj_w, wqB, wpB);
  qkv_kernel <<<BW,             256, 0, stream>>>(x, wqB, qkv_b, qB, kB, vB);
  attn_kernel<<<BW * HEADS / 4, 128, 0, stream>>>(qB, kB, vB, rel, aB);
  proj_kernel<<<BW,             256, 0, stream>>>(aB, wpB, proj_b, (float*)d_out);
}
